// LSTMSequentialEncoder_13340168422050
// MI455X (gfx1250) — compile-verified, matrix operand order checked by reading, not run on hardware
//
#include <hip/hip_runtime.h>
#include <hip/hip_bf16.h>
#include <math.h>

#define BATCH   32
#define TSTEPS  24
#define DIM     13
#define HW      32
#define NPIX    1024      // 32*32
#define C2      26        // 2*DIM
#define G4      52        // 4*DIM
#define KIM     234       // 26*9 im2col depth
#define KPAD    256
#define NCLS    8

typedef _Float16 half8 __attribute__((ext_vector_type(8)));
typedef _Float16 v16h  __attribute__((ext_vector_type(16)));
typedef float    v8f   __attribute__((ext_vector_type(8)));
typedef unsigned int v4u __attribute__((ext_vector_type(4)));
typedef int          v4i __attribute__((ext_vector_type(4)));
typedef int          v8i __attribute__((ext_vector_type(8)));

// Build one 16-half WMMA fragment from two 16-byte chunks.
__device__ __forceinline__ v16h pack16(const _Float16* p0, const _Float16* p1) {
    union { v16h v; half8 h[2]; } u;
    u.h[0] = *(const half8*)p0;
    u.h[1] = *(const half8*)p1;
    return u.v;
}

__device__ __forceinline__ float sigf(float x) { return 1.0f / (1.0f + expf(-x)); }

// ---------------------------------------------------------------------------
// Tensor Data Mover: 1-D contiguous tile of `nelem` f16 elements -> LDS.
// D# layout per CDNA5 ISA ch.8 (group0: count/lds/global/type, group1:
// data_size=2B, tensor_dim0=tile_dim0=nelem, stride0=nelem, 2D fields unused).
__device__ __forceinline__ void tdm_load_f16(unsigned lds_off, const void* gptr,
                                             unsigned nelem) {
    unsigned long long ga = (unsigned long long)gptr;
    v4u g0 = { 1u,                                   // count=1, user desc
               lds_off,                              // lds_addr (bytes)
               (unsigned)ga,                         // global_addr[31:0]
               ((unsigned)(ga >> 32) & 0x01FFFFFFu)  // global_addr[56:32]
                   | 0x80000000u };                  // type=2 ("image")
    v8i g1 = { (int)(1u << 16),                      // data_size=1 (2 bytes)
               (int)((nelem & 0xFFFFu) << 16),       // tensor_dim0[15:0]
               (int)((nelem >> 16) & 0xFFFFu),       // tensor_dim0[31:16], dim1=0
               (int)((nelem & 0xFFFFu) << 16),       // tile_dim0
               0,                                    // tile_dim1/2 unused
               (int)nelem,                           // tensor_dim0_stride[31:0]
               0, 0 };
    v4i gz = { 0, 0, 0, 0 };
#if __clang_major__ >= 23
    v8i gz8 = { 0, 0, 0, 0, 0, 0, 0, 0 };
    __builtin_amdgcn_tensor_load_to_lds(g0, g1, gz, gz, gz8, 0);
#else
    __builtin_amdgcn_tensor_load_to_lds(g0, g1, gz, gz, 0);
#endif
}

// ---------------------------------------------------------------------------
__global__ void zero_kernel(float* p, int n) {
    for (int i = blockIdx.x * blockDim.x + threadIdx.x; i < n; i += gridDim.x * blockDim.x)
        p[i] = 0.0f;
}

// W_gates [52][26][3][3] f32 -> padded f16 [64][256], row-major, k = ci*9+kh*3+kw
__global__ void wg_prep_kernel(const float* __restrict__ Wg, _Float16* __restrict__ wg16) {
    int o = blockIdx.x;       // 0..63
    int k = threadIdx.x;      // 0..255
    float v = 0.0f;
    if (o < G4 && k < KIM) {
        int c = k / 9, rem = k % 9;
        v = Wg[(o * C2 + c) * 9 + rem];
    }
    wg16[o * KPAD + k] = (_Float16)v;
}

// ---------------------------------------------------------------------------
// ConvLSTM gates via WMMA implicit GEMM (M=64 pad of 52, K=256 pad of 234,
// N = 16 pixels per wave) + fused LSTM pointwise.
__global__ void __launch_bounds__(32)
conv_lstm_kernel(const float* __restrict__ x1, const _Float16* __restrict__ wg16,
                 const float* __restrict__ hbuf, float* __restrict__ cbuf,
                 float* __restrict__ htmp, int t)
{
    __shared__ _Float16 patch[16][KPAD];   // im2col tile (B operand, row per pixel)
    __shared__ float    gtile[64][16];     // gate outputs

    const int lane = threadIdx.x;
    const int tile = blockIdx.x;           // 16-pixel tile index (0..63)
    const int b    = blockIdx.y;

    const float* xt = x1 + ((size_t)(b * TSTEPS + t) * DIM) * NPIX;
    const float* hb = hbuf + (size_t)b * DIM * NPIX;

    // im2col with SAME-padding (zeros); k >= KIM padded with zeros
    for (int idx = lane; idx < 16 * KPAD; idx += 32) {
        int p = idx >> 8;
        int k = idx & (KPAD - 1);
        float v = 0.0f;
        if (k < KIM) {
            int c = k / 9, rem = k % 9;
            int dy = rem / 3, dx = rem % 3;
            int pix = tile * 16 + p;
            int y = (pix >> 5) + dy - 1;
            int x = (pix & 31) + dx - 1;
            if (y >= 0 && y < HW && x >= 0 && x < HW)
                v = (c < DIM) ? xt[c * NPIX + y * HW + x]
                              : hb[(c - DIM) * NPIX + y * HW + x];
        }
        patch[p][k] = (_Float16)v;
    }
    __syncthreads();

    const int row16 = lane & 15;
    const int kOff  = (lane < 16) ? 0 : 8;   // f16 A/B fragment lane layout
    const int hi    = (lane >> 4) * 8;       // f32 D fragment row offset

    for (int mt = 0; mt < 4; ++mt) {
        v8f acc = {};
        for (int kt = 0; kt < KPAD / 32; ++kt) {
            const int kbase = kt * 32;
            const _Float16* aw = wg16 + (size_t)(mt * 16 + row16) * KPAD + kbase + kOff;
            v16h a = pack16(aw, aw + 16);
            const _Float16* bp = &patch[row16][kbase + kOff];
            v16h bm = pack16(bp, bp + 16);
            acc = __builtin_amdgcn_wmma_f32_16x16x32_f16(false, a, false, bm,
                                                         (short)0, acc, false, false);
        }
        for (int r = 0; r < 8; ++r)
            gtile[mt * 16 + r + hi][row16] = acc[r];
    }
    __syncthreads();

    // LSTM pointwise: gates order i,f,g,o (13 channels each)
    for (int j = lane; j < DIM * 16; j += 32) {
        int ch = j >> 4;
        int px = j & 15;
        int n  = tile * 16 + px;
        float ig = gtile[ch][px];
        float fg = gtile[ch + DIM][px];
        float gg = gtile[ch + 2 * DIM][px];
        float og = gtile[ch + 3 * DIM][px];
        size_t off = ((size_t)b * DIM + ch) * NPIX + n;
        float cn = sigf(fg) * cbuf[off] + sigf(ig) * tanhf(gg);
        cbuf[off] = cn;
        htmp[off] = sigf(og) * tanhf(cn);
    }
}

// ---------------------------------------------------------------------------
// 13x13 projections -> padded f16 operands for the attention GEMMs.
// q/kh/km stored [b][n][32] (per-token, d padded to 32: A/B fragment friendly).
// vh/vm stored  [b][16][N] (row-major d x N: A fragment for Z = V*P).
__global__ void __launch_bounds__(256)
proj_kernel(const float* __restrict__ htmp, const float* __restrict__ mbuf,
            const float* __restrict__ Wq,  const float* __restrict__ Wkh,
            const float* __restrict__ Wvh, const float* __restrict__ Wkm,
            const float* __restrict__ Wvm,
            _Float16* __restrict__ qf,  _Float16* __restrict__ khf,
            _Float16* __restrict__ kmf, _Float16* __restrict__ vhf,
            _Float16* __restrict__ vmf)
{
    __shared__ float sw[5 * DIM * DIM];
    for (int i = threadIdx.x; i < DIM * DIM; i += 256) {
        sw[i]       = Wq[i];
        sw[i + 169] = Wkh[i];
        sw[i + 338] = Wvh[i];
        sw[i + 507] = Wkm[i];
        sw[i + 676] = Wvm[i];
    }
    __syncthreads();

    int gid = blockIdx.x * 256 + threadIdx.x;  // b*NPIX + n
    int b = gid >> 10;
    int n = gid & (NPIX - 1);

    float hv[DIM], mv[DIM];
    size_t base = (size_t)b * DIM * NPIX + n;
    for (int d = 0; d < DIM; ++d) {
        hv[d] = htmp[base + d * NPIX];
        mv[d] = mbuf[base + d * NPIX];
    }
    size_t rrow = ((size_t)b * NPIX + n) * 32;
    for (int o = 0; o < 32; ++o) {
        float aq = 0.f, akh = 0.f, akm = 0.f, avh = 0.f, avm = 0.f;
        if (o < DIM) {
            for (int d = 0; d < DIM; ++d) {
                float h = hv[d], m = mv[d];
                aq  += sw[o * DIM + d] * h;
                akh += sw[169 + o * DIM + d] * h;
                avh += sw[338 + o * DIM + d] * h;
                akm += sw[507 + o * DIM + d] * m;
                avm += sw[676 + o * DIM + d] * m;
            }
        }
        qf [rrow + o] = (_Float16)aq;
        khf[rrow + o] = (_Float16)akh;
        kmf[rrow + o] = (_Float16)akm;
        if (o < 16) {
            vhf[((size_t)b * 16 + o) * NPIX + n] = (_Float16)avh;
            vmf[((size_t)b * 16 + o) * NPIX + n] = (_Float16)avm;
        }
    }
}

// ---------------------------------------------------------------------------
// Full 1024x1024 spatial attention. 4 waves per block, each wave owns a
// 16-query tile; K (64KB) and V (32KB) for the whole batch are staged into
// LDS ONCE per block by the Tensor Data Mover (wave 0, TENSORcnt-tracked),
// then all GEMM operand traffic is ds_load_b128. 96 WMMAs per wave.
// Dynamic LDS: K 64KB | V 32KB | 4 x 32KB score tiles = 224KB of WGP pool.
// blockIdx.z: 0 -> (kh,vh)->zh ; 1 -> (km,vm)->zm.
#define ATTN_LDS_K   0
#define ATTN_LDS_V   65536
#define ATTN_LDS_S   98304
#define ATTN_LDS_TOT (ATTN_LDS_S + 4 * 32768)

__global__ void __launch_bounds__(128)
attn_kernel(const _Float16* __restrict__ qf,
            const _Float16* __restrict__ khf, const _Float16* __restrict__ vhf,
            const _Float16* __restrict__ kmf, const _Float16* __restrict__ vmf,
            float* __restrict__ zh, float* __restrict__ zm)
{
    extern __shared__ char smem[];
    _Float16* kS = (_Float16*)(smem + ATTN_LDS_K);   // [1024][32] keys (d padded)
    _Float16* vS = (_Float16*)(smem + ATTN_LDS_V);   // [16][1024] values

    const int lane = threadIdx.x & 31;
    const int wid  = threadIdx.x >> 5;
    const int qt   = blockIdx.x * 4 + wid;            // 16-query tile per wave
    const int b    = blockIdx.y;
    const _Float16* kf = blockIdx.z ? kmf : khf;
    const _Float16* vf = blockIdx.z ? vmf : vhf;
    float*          zf = blockIdx.z ? zm  : zh;

    _Float16* sT = (_Float16*)(smem + ATTN_LDS_S + wid * 32768);  // [16][1024]

    // TDM: stage this batch's K and V tiles into LDS (wave 0 issues; DMA runs
    // asynchronously, tracked by TENSORcnt).
    if (threadIdx.x < 32) {
        tdm_load_f16(ATTN_LDS_K, kf + (size_t)b * NPIX * 32, NPIX * 32);
        tdm_load_f16(ATTN_LDS_V, vf + (size_t)b * 16 * NPIX, 16 * NPIX);
        __builtin_amdgcn_s_wait_tensorcnt(0);
    }
    __syncthreads();

    const int row16 = lane & 15;
    const int kOff  = (lane < 16) ? 0 : 8;
    const int hi    = (lane >> 4) * 8;

    // Q tile (B operand): 16 queries x K=32 (d padded), loaded once from global
    const _Float16* qp = qf + ((size_t)b * NPIX + qt * 16 + row16) * 32 + kOff;
    const v16h qv = pack16(qp, qp + 16);

    // S = K^T Q  (M = 16 keys per step, 64 steps; A fragments from LDS)
    for (int kb = 0; kb < NPIX / 16; ++kb) {
        const _Float16* kp = kS + (kb * 16 + row16) * 32 + kOff;
        v16h kv = pack16(kp, kp + 16);
        v8f s = {};
        s = __builtin_amdgcn_wmma_f32_16x16x32_f16(false, kv, false, qv,
                                                   (short)0, s, false, false);
        for (int r = 0; r < 8; ++r)
            sT[row16 * NPIX + kb * 16 + r + hi] = (_Float16)s[r];  // sT[query][key]
    }
    __syncthreads();

    // Row softmax over keys, all 32 lanes: 2 lanes per query (512 keys each),
    // cross-half combine via shfl_xor(16). Reference uses un-scaled q.k.
    {
        _Float16* srow = sT + row16 * NPIX;
        const int k0 = (lane >> 4) * 512;
        float mx = -1e30f;
        for (int k = k0; k < k0 + 512; ++k) mx = fmaxf(mx, (float)srow[k]);
        mx = fmaxf(mx, __shfl_xor(mx, 16, 32));
        float sum = 0.f;
        for (int k = k0; k < k0 + 512; ++k) sum += expf((float)srow[k] - mx);
        sum += __shfl_xor(sum, 16, 32);
        float inv = 1.f / sum;
        for (int k = k0; k < k0 + 512; ++k)
            srow[k] = (_Float16)(expf((float)srow[k] - mx) * inv);
    }
    __syncthreads();

    // Z = V * P  (K = 32 keys per step, 32 steps; A = V from LDS, B = probs)
    v8f acc = {};
    for (int kb = 0; kb < NPIX / 32; ++kb) {
        const _Float16* vp = vS + row16 * NPIX + kb * 32 + kOff;
        v16h av = pack16(vp, vp + 16);
        const _Float16* pp = sT + row16 * NPIX + kb * 32 + kOff;
        v16h bv = pack16(pp, pp + 16);
        acc = __builtin_amdgcn_wmma_f32_16x16x32_f16(false, av, false, bv,
                                                     (short)0, acc, false, false);
    }
    for (int r = 0; r < 8; ++r)
        zf[((size_t)b * 16 + r + hi) * NPIX + qt * 16 + row16] = acc[r];
}

// ---------------------------------------------------------------------------
// SAM fuse + memory/hidden update (per spatial token)
__global__ void __launch_bounds__(256)
sam_update_kernel(const float* __restrict__ zh, const float* __restrict__ zm,
                  const float* __restrict__ htmp,
                  const float* __restrict__ Wz, const float* __restrict__ Wm,
                  float* __restrict__ hbuf, float* __restrict__ mbuf)
{
    __shared__ float swz[DIM * C2];
    __shared__ float swm[3 * DIM * C2];
    for (int i = threadIdx.x; i < DIM * C2; i += 256)     swz[i] = Wz[i];
    for (int i = threadIdx.x; i < 3 * DIM * C2; i += 256) swm[i] = Wm[i];
    __syncthreads();

    int gid = blockIdx.x * 256 + threadIdx.x;
    int b = gid >> 10, n = gid & (NPIX - 1);

    float zin[C2], hv[DIM], mv[DIM];
    size_t base13 = (size_t)b * DIM * NPIX + n;
    size_t base16 = (size_t)b * 16  * NPIX + n;
    for (int d = 0; d < DIM; ++d) {
        zin[d]       = zh[base16 + d * NPIX];
        zin[d + DIM] = zm[base16 + d * NPIX];
        hv[d]        = htmp[base13 + d * NPIX];
        mv[d]        = mbuf[base13 + d * NPIX];
    }
    float cat[C2];
    for (int o = 0; o < DIM; ++o) {
        float a = 0.f;
        for (int c = 0; c < C2; ++c) a += swz[o * C2 + c] * zin[c];
        cat[o] = a;                    // Z
    }
    for (int d = 0; d < DIM; ++d) cat[d + DIM] = hv[d];   // [Z ; h]

    for (int o = 0; o < DIM; ++o) {
        float mo = 0.f, mg = 0.f, mi = 0.f;
        for (int c = 0; c < C2; ++c) {
            mo += swm[o * C2 + c] * cat[c];
            mg += swm[(o + DIM) * C2 + c] * cat[c];
            mi += swm[(o + 2 * DIM) * C2 + c] * cat[c];
        }
        float si = sigf(mi);
        float mn = (1.f - si) * mv[o] + si * tanhf(mg);
        mbuf[base13 + o * NPIX] = mn;
        hbuf[base13 + o * NPIX] = sigf(mo) * mn;
    }
}

// ---------------------------------------------------------------------------
// 1x1 conv on final cell state + log_softmax over 8 classes
__global__ void __launch_bounds__(256)
final_kernel(const float* __restrict__ cbuf, const float* __restrict__ Wf,
             const float* __restrict__ bf, float* __restrict__ out)
{
    __shared__ float sw[NCLS * DIM];
    __shared__ float sb[NCLS];
    for (int i = threadIdx.x; i < NCLS * DIM; i += 256) sw[i] = Wf[i];
    if (threadIdx.x < NCLS) sb[threadIdx.x] = bf[threadIdx.x];
    __syncthreads();

    int gid = blockIdx.x * 256 + threadIdx.x;
    int b = gid >> 10, n = gid & (NPIX - 1);

    float cv[DIM];
    size_t base = (size_t)b * DIM * NPIX + n;
    for (int d = 0; d < DIM; ++d) cv[d] = cbuf[base + d * NPIX];

    float lg[NCLS];
    float mx = -1e30f;
    for (int k = 0; k < NCLS; ++k) {
        float a = sb[k];
        for (int d = 0; d < DIM; ++d) a += sw[k * DIM + d] * cv[d];
        lg[k] = a;
        mx = fmaxf(mx, a);
    }
    float s = 0.f;
    for (int k = 0; k < NCLS; ++k) s += expf(lg[k] - mx);
    float lse = mx + logf(s);
    for (int k = 0; k < NCLS; ++k)
        out[((size_t)b * NCLS + k) * NPIX + n] = lg[k] - lse;
}

// ---------------------------------------------------------------------------
extern "C" void kernel_launch(void* const* d_in, const int* in_sizes, int n_in,
                              void* d_out, int out_size, void* d_ws, size_t ws_size,
                              hipStream_t stream)
{
    (void)in_sizes; (void)n_in; (void)out_size; (void)ws_size;

    const float* x1  = (const float*)d_in[0];
    const float* Wg  = (const float*)d_in[1];
    const float* Wq  = (const float*)d_in[2];
    const float* Wkh = (const float*)d_in[3];
    const float* Wvh = (const float*)d_in[4];
    const float* Wkm = (const float*)d_in[5];
    const float* Wvm = (const float*)d_in[6];
    const float* Wz  = (const float*)d_in[7];
    const float* Wm  = (const float*)d_in[8];
    const float* Wf  = (const float*)d_in[9];
    const float* bf  = (const float*)d_in[10];

    char* ws = (char*)d_ws;
    size_t cur = 0;
    auto alloc = [&](size_t bytes) -> void* {
        void* p = ws + cur;
        cur = (cur + bytes + 255) & ~(size_t)255;
        return p;
    };

    const size_t stateF = (size_t)BATCH * DIM * NPIX;       // f32 elements
    float*     cbuf = (float*)alloc(stateF * sizeof(float));
    float*     hbuf = (float*)alloc(stateF * sizeof(float));
    float*     mbuf = (float*)alloc(stateF * sizeof(float));
    float*     htmp = (float*)alloc(stateF * sizeof(float));
    float*     zhf  = (float*)alloc((size_t)BATCH * 16 * NPIX * sizeof(float));
    float*     zmf  = (float*)alloc((size_t)BATCH * 16 * NPIX * sizeof(float));
    _Float16*  qf   = (_Float16*)alloc((size_t)BATCH * NPIX * 32 * sizeof(_Float16));
    _Float16*  khf  = (_Float16*)alloc((size_t)BATCH * NPIX * 32 * sizeof(_Float16));
    _Float16*  kmf  = (_Float16*)alloc((size_t)BATCH * NPIX * 32 * sizeof(_Float16));
    _Float16*  vhf  = (_Float16*)alloc((size_t)BATCH * 16 * NPIX * sizeof(_Float16));
    _Float16*  vmf  = (_Float16*)alloc((size_t)BATCH * 16 * NPIX * sizeof(_Float16));
    _Float16*  wg16 = (_Float16*)alloc((size_t)64 * KPAD * sizeof(_Float16));

    // c, h, m are allocated contiguously (sizes are 256B multiples): zero all three
    zero_kernel<<<dim3(1024), dim3(256), 0, stream>>>(cbuf, (int)(3 * stateF));
    wg_prep_kernel<<<dim3(64), dim3(256), 0, stream>>>(Wg, wg16);

    for (int t = 0; t < TSTEPS; ++t) {
        conv_lstm_kernel<<<dim3(NPIX / 16, BATCH), dim3(32), 0, stream>>>(
            x1, wg16, hbuf, cbuf, htmp, t);
        proj_kernel<<<dim3(BATCH * NPIX / 256), dim3(256), 0, stream>>>(
            htmp, mbuf, Wq, Wkh, Wvh, Wkm, Wvm, qf, khf, kmf, vhf, vmf);
        attn_kernel<<<dim3(NPIX / 64, BATCH, 2), dim3(128), ATTN_LDS_TOT, stream>>>(
            qf, khf, vhf, kmf, vmf, zhf, zmf);
        sam_update_kernel<<<dim3(BATCH * NPIX / 256), dim3(256), 0, stream>>>(
            zhf, zmf, htmp, Wz, Wm, hbuf, mbuf);
    }

    final_kernel<<<dim3(BATCH * NPIX / 256), dim3(256), 0, stream>>>(
        cbuf, Wf, bf, (float*)d_out);
}
